// CrossAttention_60318520705418
// MI455X (gfx1250) — matrix operand order repaired and checked by reading, not compiled, not run on hardware
//
#include <hip/hip_runtime.h>

typedef _Float16 half_t;
typedef __attribute__((ext_vector_type(2)))  __fp16   v2fp16;
typedef __attribute__((ext_vector_type(16))) _Float16 v16h;
typedef __attribute__((ext_vector_type(8)))  float    v8f;
typedef __attribute__((ext_vector_type(4)))  unsigned u32x4_t;
typedef __attribute__((ext_vector_type(4)))  int      i32x4_t;
typedef __attribute__((ext_vector_type(8)))  int      i32x8_t;

constexpr int B_  = 2;
constexpr int SQ  = 2048;
constexpr int SK  = 2048;
constexpr int H_  = 16;
constexpr int HKV = 8;
constexpr int D_  = 128;

constexpr int QB = 128;  // query rows per workgroup (8 waves x 16)
constexpr int KB = 64;   // key block per iteration
constexpr float SCALE = 0.08838834764831845f; // 1/sqrt(128)

// LDS map (112KB total). Qs (prologue-only) aliases Ks+VsT.
constexpr unsigned OFF_KF32 = 0;                 // f32 [64][128]  32KB  (TDM landing)
constexpr unsigned OFF_VF32 = 32768;             // f32 [64][128]  32KB  (TDM landing)
constexpr unsigned OFF_KS   = 65536;             // u32 [64][64]   16KB  packed f16 pairs
constexpr unsigned OFF_VST  = 81920;             // f16 [128][64]  16KB  transposed V
constexpr unsigned OFF_PS   = 98304;             // f16 [128][64]  16KB  exp(S)
constexpr unsigned SMEM_BYTES = 114688;

union FragH { v16h v; unsigned u[8]; };
union FragF { v8f  v; float    f[8]; };

static __device__ inline unsigned pack2(float a, float b) {
  union { v2fp16 h; unsigned u; } x;
  x.h = __builtin_amdgcn_cvt_pkrtz(a, b);   // v_cvt_pk_f16_f32 (RTZ)
  return x.u;
}

// TDM: DMA a 2D f32 tile (tile_w x tile_h, row stride in elements) from global
// into LDS at byte offset lds_off, rows packed contiguously. Tracked on TENSORcnt.
static __device__ inline void tdm_load_2d_f32(unsigned lds_off, const float* g,
                                              unsigned tile_w, unsigned tile_h,
                                              unsigned long long stride_elems)
{
  unsigned long long ga = (unsigned long long)(const void*)g;
  u32x4_t g0;
  g0[0] = 1u;                                               // count=1 (valid user D#)
  g0[1] = lds_off;                                          // lds_addr [63:32]
  g0[2] = (unsigned)ga;                                     // global_addr[31:0]
  g0[3] = (unsigned)((ga >> 32) & 0x1FFFFFFull) | (2u << 30); // addr[56:32] | type=2
  i32x8_t g1;
  g1[0] = (int)(2u << 16);                                  // data_size=2 (4 bytes)
  g1[1] = (int)((tile_w & 0xFFFFu) << 16);                  // tensor_dim0 lo16
  g1[2] = (int)((tile_w >> 16) | ((tile_h & 0xFFFFu) << 16)); // dim0 hi | dim1 lo
  g1[3] = (int)((tile_h >> 16) | ((tile_w & 0xFFFFu) << 16)); // dim1 hi | tile_dim0
  g1[4] = (int)(tile_h & 0xFFFFu);                          // tile_dim1 | tile_dim2=0
  g1[5] = (int)(stride_elems & 0xFFFFFFFFull);              // tensor_dim0_stride lo
  g1[6] = (int)((stride_elems >> 32) & 0xFFFFull);          // stride hi | dim1_stride lo=0
  g1[7] = 0;
  i32x4_t z4 = {0, 0, 0, 0};
#if defined(__clang_major__) && (__clang_major__ >= 23)
  i32x8_t z8 = {0, 0, 0, 0, 0, 0, 0, 0};
  __builtin_amdgcn_tensor_load_to_lds(g0, g1, z4, z4, z8, 0);
#else
  __builtin_amdgcn_tensor_load_to_lds(g0, g1, z4, z4, 0);
#endif
}

__global__ __launch_bounds__(256)
void fa_fwd_kernel(const float* __restrict__ q,
                   const float* __restrict__ kv,
                   float* __restrict__ out)
{
  __shared__ unsigned char smem[SMEM_BYTES];
  float*    Kf32 = (float*)(smem + OFF_KF32);
  float*    Vf32 = (float*)(smem + OFF_VF32);
  unsigned* Ks   = (unsigned*)(smem + OFF_KS);   // [key][d/2] packed f16 pairs
  half_t*   VsT  = (half_t*)(smem + OFF_VST);    // [d][key]
  half_t*   Ps   = (half_t*)(smem + OFF_PS);     // [row][key]
  unsigned* Qs   = (unsigned*)(smem + OFF_KS);   // alias (prologue only)

  const unsigned lds_base = (unsigned)(unsigned long long)(void*)smem;

  const int b   = blockIdx.z;
  const int h   = blockIdx.y;
  const int hk  = h / (H_ / HKV);
  const int q0  = blockIdx.x * QB;

  const int tid  = threadIdx.x;
  const int wave = tid >> 5;
  const int lane = tid & 31;
  const int l16  = lane & 15;
  const int hi   = lane >> 4;

  const int kb_end = (q0 + QB) / KB;  // causal: key blocks with k0 <= q0+QB-1
  const unsigned long long kv_row_stride = 2ull * HKV * D_;   // elems between keys

  // kv base for (b, key=k0, k/v, hk, 0)
  auto kv_ptr = [&](int k0, int which) -> const float* {
    return kv + (((size_t)b * SK + k0) * 2 + which) * (size_t)(HKV * D_) + (size_t)hk * D_;
  };

  // ---- prologue: kick off TDM for block 0 (overlaps Q staging) ----
  if (wave == 0) {
    tdm_load_2d_f32(lds_base + OFF_KF32, kv_ptr(0, 0), D_, KB, kv_row_stride);
    tdm_load_2d_f32(lds_base + OFF_VF32, kv_ptr(0, 1), D_, KB, kv_row_stride);
  }

  // ---- stage Q block (128 x 128 fp32 -> f16) into LDS ----
  for (int i = tid; i < QB * (D_ / 4); i += 256) {
    int row = i >> 5;
    int dv  = i & 31;
    const float4 f = *(const float4*)(q + (((size_t)b * SQ + (q0 + row)) * H_ + h) * D_ + dv * 4);
    Qs[row * (D_ / 2) + dv * 2 + 0] = pack2(f.x, f.y);
    Qs[row * (D_ / 2) + dv * 2 + 1] = pack2(f.z, f.w);
  }
  __syncthreads();

  // ---- capture Q A-fragments: row M = lane%16; d = 32c+(j/4)*16+hi*8+(j%4)*2 ----
  FragH aQ[4];
  {
    const int row = wave * 16 + l16;
    for (int c = 0; c < 4; ++c)
      for (int j = 0; j < 8; ++j)
        aQ[c].u[j] = Qs[row * (D_ / 2) + 16 * c + (j >> 2) * 8 + hi * 4 + (j & 3)];
  }

  v8f acc[8];
  for (int i = 0; i < 8; ++i) acc[i] = {};
  float mrow[8], lrow[8];
  for (int r = 0; r < 8; ++r) { mrow[r] = -3.0e38f; lrow[r] = 0.0f; }

  for (int jb = 0; jb < kb_end; ++jb) {
    const int k0 = jb * KB;

    __builtin_amdgcn_s_wait_tensorcnt((short)0);  // block jb landed (wave0; no-op others)
    __syncthreads();                              // + fences Qs/Ks alias & buffer reuse

    // ---- convert pass: Kf32 -> packed f16 Ks, Vf32 -> transposed f16 VsT ----
    for (int i = tid; i < KB * (D_ / 4); i += 256) {
      int row = i >> 5;
      int dv  = i & 31;
      float4 kf = ((const float4*)Kf32)[i];
      float4 vf = ((const float4*)Vf32)[i];
      Ks[row * (D_ / 2) + dv * 2 + 0] = pack2(kf.x, kf.y);
      Ks[row * (D_ / 2) + dv * 2 + 1] = pack2(kf.z, kf.w);
      VsT[(dv * 4 + 0) * KB + row] = (half_t)vf.x;
      VsT[(dv * 4 + 1) * KB + row] = (half_t)vf.y;
      VsT[(dv * 4 + 2) * KB + row] = (half_t)vf.z;
      VsT[(dv * 4 + 3) * KB + row] = (half_t)vf.w;
    }
    __syncthreads();

    // ---- kick off TDM for block jb+1; overlaps S/softmax/PV below ----
    if (wave == 0 && jb + 1 < kb_end) {
      tdm_load_2d_f32(lds_base + OFF_KF32, kv_ptr(k0 + KB, 0), D_, KB, kv_row_stride);
      tdm_load_2d_f32(lds_base + OFF_VF32, kv_ptr(k0 + KB, 1), D_, KB, kv_row_stride);
    }

    // ---- S = Q K^T : 16 rows x 64 keys (4 N-tiles; batch B-frags per tile) ----
    FragF s[4];
    for (int t = 0; t < 4; ++t) {
      const int key = 16 * t + l16;           // B layout: N = lane%16
      FragH bK[4];
      for (int c = 0; c < 4; ++c)
        for (int j = 0; j < 8; ++j)
          bK[c].u[j] = Ks[key * (D_ / 2) + 16 * c + (j >> 2) * 8 + hi * 4 + (j & 3)];
      v8f c_ = {};
      for (int c = 0; c < 4; ++c)
        c_ = __builtin_amdgcn_wmma_f32_16x16x32_f16(false, aQ[c].v, false, bK[c].v,
                                                    (short)0, c_, false, false);
      s[t].v = c_;
    }

    // ---- scale + causal mask + online softmax ----
    float mb[8];
    for (int r = 0; r < 8; ++r) mb[r] = -3.0e38f;
    for (int t = 0; t < 4; ++t) {
      const int col = k0 + 16 * t + l16;
      for (int r = 0; r < 8; ++r) {
        const int row = q0 + wave * 16 + r + 8 * hi;
        float sv = s[t].f[r] * SCALE;
        sv = (col > row) ? -3.0e38f : sv;
        s[t].f[r] = sv;
        mb[r] = fmaxf(mb[r], sv);
      }
    }
    for (int r = 0; r < 8; ++r) {
      for (int off = 1; off < 16; off <<= 1)
        mb[r] = fmaxf(mb[r], __shfl_xor(mb[r], off, 32));
      const float mnew  = fmaxf(mrow[r], mb[r]);
      const float alpha = __expf(mrow[r] - mnew);
      mrow[r] = mnew;
      lrow[r] *= alpha;
      for (int ct = 0; ct < 8; ++ct) acc[ct][r] *= alpha;
    }

    float rs[8];
    for (int r = 0; r < 8; ++r) rs[r] = 0.0f;
    for (int t = 0; t < 4; ++t)
      for (int r = 0; r < 8; ++r) {
        const float p = __expf(s[t].f[r] - mrow[r]);
        rs[r] += p;
        const int row = wave * 16 + r + 8 * hi;
        const int key = 16 * t + l16;
        Ps[row * KB + key] = (half_t)p;
      }
    for (int r = 0; r < 8; ++r) {
      for (int off = 1; off < 16; off <<= 1)
        rs[r] += __shfl_xor(rs[r], off, 32);
      lrow[r] += rs[r];
    }
    __syncthreads();

    // ---- O += P V : 2 key-chunks x 8 d-tiles ----
    for (int kk = 0; kk < 2; ++kk) {
      FragH aP;
      const int row = wave * 16 + l16;
      for (int j = 0; j < 8; ++j) {
        const int key = 32 * kk + (j >> 2) * 16 + hi * 8 + (j & 3) * 2;
        aP.u[j] = *(const unsigned*)&Ps[row * KB + key];
      }
      for (int ct = 0; ct < 8; ++ct) {
        FragH bV;
        const int d = 16 * ct + l16;
        for (int j = 0; j < 8; ++j) {
          const int key = 32 * kk + (j >> 2) * 16 + hi * 8 + (j & 3) * 2;
          bV.u[j] = *(const unsigned*)&VsT[d * KB + key];
        }
        acc[ct] = __builtin_amdgcn_wmma_f32_16x16x32_f16(false, aP.v, false, bV.v,
                                                         (short)0, acc[ct], false, false);
      }
    }
  }

  // ---- normalize and write out ----
  for (int ct = 0; ct < 8; ++ct)
    for (int r = 0; r < 8; ++r) {
      const int row = q0 + wave * 16 + r + 8 * hi;
      const int d   = 16 * ct + l16;
      out[(((size_t)b * SQ + row) * H_ + h) * D_ + d] = acc[ct][r] / lrow[r];
    }
}

extern "C" void kernel_launch(void* const* d_in, const int* in_sizes, int n_in,
                              void* d_out, int out_size, void* d_ws, size_t ws_size,
                              hipStream_t stream) {
  (void)in_sizes; (void)n_in; (void)out_size; (void)d_ws; (void)ws_size;
  const float* q  = (const float*)d_in[0];
  const float* kv = (const float*)d_in[1];
  float* out = (float*)d_out;
  dim3 grid(SQ / QB, H_, B_);
  fa_fwd_kernel<<<grid, 256, 0, stream>>>(q, kv, out);
}
